// GATBlock_39444979646658
// MI455X (gfx1250) — compile-verified, matrix operand order checked by reading, not run on hardware
//
#include <hip/hip_runtime.h>
#include <hip/hip_bf16.h>
#include <math.h>

// ---------------------------------------------------------------------------
// GATv2 block for MI455X (gfx1250, wave32).
// GEMMs use v_wmma_f32_16x16x32_f16 (f16 in, f32 accumulate).
// Edge phase: wave-per-edge, 4 heads x 8 lanes x 8 floats, L2-resident scatter.
// ---------------------------------------------------------------------------

typedef __attribute__((ext_vector_type(16))) _Float16 v16h;
typedef __attribute__((ext_vector_type(8)))  _Float16 v8h;
typedef __attribute__((ext_vector_type(8)))  float    v8f;

#define NN    20000
#define ERAW  320000
#define ETOT  (ERAW + NN)   // 340000 incl. self loops
#define IND   64
#define OUTD  64
#define HEADS 4
#define HD    256           // HEADS*OUTD
#define NEG   0.2f

// ----- float atomic max via CAS (portable, lowers to global_atomic_cmpswap) -
__device__ inline void atomicMaxF(float* addr, float val) {
  unsigned* ua = (unsigned*)addr;
  unsigned cur = *ua;
  while (__uint_as_float(cur) < val) {
    unsigned prev = atomicCAS(ua, cur, __float_as_uint(val));
    if (prev == cur) break;
    cur = prev;
  }
}

// ----- init: zero aggregation buffers, -inf node max, zero reduce scratch ---
__global__ void k_init(float* __restrict__ hagg, float* __restrict__ den,
                       float* __restrict__ nodemax, float* __restrict__ eabuf) {
  long long t = (long long)blockIdx.x * 256 + threadIdx.x;
  if (t < (long long)NN * HD) hagg[t] = 0.f;
  if (t < NN * HEADS) { den[t] = 0.f; nodemax[t] = -3.402823466e38f; }
  if (t < 516) eabuf[t] = 0.f;
}

// ----- deterministic mean of edge_attr (EDGE_DIM==1) ------------------------
__global__ void k_ea1(const float* __restrict__ ea, float* __restrict__ part) {
  __shared__ float red[256];
  int t = threadIdx.x;
  float s = 0.f;
  for (int i = blockIdx.x * 256 + t; i < ERAW; i += 512 * 256) s += ea[i];
  red[t] = s; __syncthreads();
  for (int m = 128; m > 0; m >>= 1) { if (t < m) red[t] += red[t + m]; __syncthreads(); }
  if (t == 0) part[blockIdx.x] = red[0];
}
__global__ void k_ea2(float* __restrict__ eabuf) {
  __shared__ float red[512];
  int t = threadIdx.x;
  red[t] = eabuf[t]; __syncthreads();
  for (int m = 256; m > 0; m >>= 1) { if (t < m) red[t] += red[t + m]; __syncthreads(); }
  if (t == 0) eabuf[512] = red[0] * (1.f / ERAW);   // slot 512 = ea_mean
}

// ----- x (f32) -> f16 row-major --------------------------------------------
__global__ void k_cvt_x(const float* __restrict__ x, _Float16* __restrict__ xf) {
  int t = blockIdx.x * 256 + threadIdx.x;
  if (t < NN * IND) xf[t] = (_Float16)x[t];
}

// ----- pre-swizzle weights into WMMA B-fragment layout ----------------------
// B fragment (16-bit, 32x16): lane = column (lane&15), K pattern per ISA
// 7.12.2: element e -> k = kb*32 + (e<8 ? half*8+e : 16+half*8+(e-8)),
// half = lane>>4. Stored so each lane loads 16 contiguous f16 (32 B).
__global__ void k_swz(const float* __restrict__ W_l, const float* __restrict__ W_r,
                      const float* __restrict__ W_p,
                      _Float16* __restrict__ blr, _Float16* __restrict__ bp) {
  int tid = blockIdx.x * 256 + threadIdx.x;
  if (tid < 32768) {               // [32 coltiles][2 kb][32 lanes][16] for W_l|W_r
    int e = tid & 15, lane = (tid >> 4) & 31, kb = (tid >> 9) & 1, t = tid >> 10;
    int nl = lane & 15, half = lane >> 4;
    int k = kb * 32 + ((e < 8) ? half * 8 + e : 16 + half * 8 + (e - 8));
    const float* W = (t < 16) ? W_l : W_r;
    blr[tid] = (_Float16)W[k * HD + (t & 15) * 16 + nl];
  } else if (tid < 49152) {        // [4 coltiles][8 kb][32 lanes][16] for W_p
    int u = tid - 32768;
    int e = u & 15, lane = (u >> 4) & 31, kb = (u >> 9) & 7, t = u >> 12;
    int nl = lane & 15, half = lane >> 4;
    int k = kb * 32 + ((e < 8) ? half * 8 + e : 16 + half * 8 + (e - 8));
    bp[u] = (_Float16)W_p[k * OUTD + t * 16 + nl];
  }
}

// ----- GEMM: x @ [W_l | W_r] + bias -> xl, xr (f32) -------------------------
// grid (1250, 8), block 128 (4 waves); wave handles one 16x16 output tile.
__global__ void k_gemm_lr(const _Float16* __restrict__ xf,
                          const _Float16* __restrict__ bswz,
                          const float* __restrict__ b_l,
                          const float* __restrict__ b_r,
                          float* __restrict__ xl, float* __restrict__ xr) {
  const int lane = threadIdx.x & 31;
  const int wave = threadIdx.x >> 5;
  const int ct = blockIdx.y * 4 + wave;     // 0..31 (0..15 -> W_l, 16..31 -> W_r)
  const int rt = blockIdx.x;                // 0..1249
  const int nl = lane & 15, half = lane >> 4;
  const _Float16* xrow = xf + (rt * 16 + nl) * IND;

  v8f c = {};
#pragma unroll
  for (int kb = 0; kb < 2; ++kb) {
    v8h g1 = *reinterpret_cast<const v8h*>(xrow + kb * 32 + half * 8);
    v8h g2 = *reinterpret_cast<const v8h*>(xrow + kb * 32 + 16 + half * 8);
    v16h a, b;
#pragma unroll
    for (int i = 0; i < 8; ++i) { a[i] = g1[i]; a[i + 8] = g2[i]; }
    b = *reinterpret_cast<const v16h*>(bswz + ((ct * 2 + kb) * 32 + lane) * 16);
    c = __builtin_amdgcn_wmma_f32_16x16x32_f16(false, a, false, b, (short)0, c,
                                               false, false);
  }
  const float bv = (ct < 16) ? b_l[ct * 16 + nl] : b_r[(ct - 16) * 16 + nl];
  float* dst = (ct < 16) ? xl : xr;
  const int colg = (ct & 15) * 16 + nl;
#pragma unroll
  for (int v = 0; v < 8; ++v) {
    int r = rt * 16 + v + (half ? 8 : 0);
    dst[r * HD + colg] = c[v] + bv;
  }
}

// ----- pass 1: logits + scatter max ----------------------------------------
__global__ void k_logit(const long long* __restrict__ ei,
                        const float* __restrict__ eattr,
                        const float* __restrict__ eamean,
                        const float* __restrict__ We,
                        const float* __restrict__ att,
                        const float* __restrict__ xl,
                        const float* __restrict__ xr,
                        float* __restrict__ elog,
                        float* __restrict__ nodemax) {
  const int lane = threadIdx.x & 31;
  const long long e = (long long)blockIdx.x * 8 + (threadIdx.x >> 5);
  if (e >= ETOT) return;
  int s, d; float ea;
  if (e < ERAW) { s = (int)ei[e]; d = (int)ei[ERAW + e]; ea = eattr[e]; }
  else          { s = d = (int)(e - ERAW); ea = eamean[0]; }
  const int head = lane >> 3;
  const int idx = head * 64 + (lane & 7) * 8;
  const float4* pl = reinterpret_cast<const float4*>(xl + (size_t)s * HD + idx);
  const float4* pr = reinterpret_cast<const float4*>(xr + (size_t)d * HD + idx);
  const float4* pw = reinterpret_cast<const float4*>(We + idx);
  const float4* pa = reinterpret_cast<const float4*>(att + idx);
  float acc = 0.f;
#pragma unroll
  for (int j = 0; j < 2; ++j) {
    float4 l = pl[j], r = pr[j], w = pw[j], at = pa[j];
    float m;
    m = l.x + r.x + ea * w.x; m = m > 0.f ? m : NEG * m; acc += m * at.x;
    m = l.y + r.y + ea * w.y; m = m > 0.f ? m : NEG * m; acc += m * at.y;
    m = l.z + r.z + ea * w.z; m = m > 0.f ? m : NEG * m; acc += m * at.z;
    m = l.w + r.w + ea * w.w; m = m > 0.f ? m : NEG * m; acc += m * at.w;
  }
  for (int m = 1; m < 8; m <<= 1) acc += __shfl_xor(acc, m, 8);
  if ((lane & 7) == 0) {
    elog[e * 4 + head] = acc;
    atomicMaxF(&nodemax[d * 4 + head], acc);
  }
}

// ----- pass 2: exp + scatter denominator -----------------------------------
__global__ void k_expsum(const long long* __restrict__ ei,
                         float* __restrict__ elog,
                         const float* __restrict__ nodemax,
                         float* __restrict__ den) {
  long long t = (long long)blockIdx.x * 256 + threadIdx.x;
  if (t >= (long long)ETOT * 4) return;
  int e = (int)(t >> 2), h = (int)(t & 3);
  int d = (e < ERAW) ? (int)ei[ERAW + e] : (e - ERAW);
  float ex = __expf(elog[t] - nodemax[d * 4 + h]);
  elog[t] = ex;
  atomicAdd(&den[d * 4 + h], ex);
}

// ----- pass 3: weighted scatter aggregation --------------------------------
__global__ void k_agg(const long long* __restrict__ ei,
                      const float* __restrict__ elog,
                      const float* __restrict__ den,
                      const float* __restrict__ xl,
                      float* __restrict__ hagg) {
  const int lane = threadIdx.x & 31;
  const long long e = (long long)blockIdx.x * 8 + (threadIdx.x >> 5);
  if (e >= ETOT) return;
  int s, d;
  if (e < ERAW) { s = (int)ei[e]; d = (int)ei[ERAW + e]; }
  else          { s = d = (int)(e - ERAW); }
  const int head = lane >> 3;
  const float alpha = elog[e * 4 + head] / (den[d * 4 + head] + 1e-16f);
  const int idx = head * 64 + (lane & 7) * 8;
  const float* ps = xl + (size_t)s * HD + idx;
  float* pd = hagg + (size_t)d * HD + idx;
#pragma unroll
  for (int i = 0; i < 8; ++i) atomicAdd(pd + i, alpha * ps[i]);
}

// ----- final: (hagg+bias_conv) @ W_p + b_p + x, ELU, LayerNorm --------------
// grid 1250, block 128 (4 waves). Each wave: 16x16 tile, K=256 (8 WMMAs).
__global__ void k_final(const float* __restrict__ hagg,
                        const float* __restrict__ bias_conv,
                        const _Float16* __restrict__ bswzp,
                        const float* __restrict__ b_p,
                        const float* __restrict__ x,
                        const float* __restrict__ gamma,
                        const float* __restrict__ beta,
                        float* __restrict__ out) {
  __shared__ float tile[16][68];
  const int lane = threadIdx.x & 31;
  const int wave = threadIdx.x >> 5;        // output col tile 0..3
  const int rt = blockIdx.x;
  const int nl = lane & 15, half = lane >> 4;
  const float* arow = hagg + (size_t)(rt * 16 + nl) * HD;

  v8f c = {};
#pragma unroll
  for (int kb = 0; kb < 8; ++kb) {
    v16h a, b;
    const int k1 = kb * 32 + half * 8;
    const int k2 = k1 + 16;
#pragma unroll
    for (int i = 0; i < 8; ++i) {
      a[i]     = (_Float16)(arow[k1 + i] + bias_conv[k1 + i]);
      a[i + 8] = (_Float16)(arow[k2 + i] + bias_conv[k2 + i]);
    }
    b = *reinterpret_cast<const v16h*>(bswzp + ((wave * 8 + kb) * 32 + lane) * 16);
    c = __builtin_amdgcn_wmma_f32_16x16x32_f16(false, a, false, b, (short)0, c,
                                               false, false);
  }
#pragma unroll
  for (int v = 0; v < 8; ++v)
    tile[v + (half ? 8 : 0)][wave * 16 + nl] = c[v];
  __syncthreads();

  const int r = threadIdx.x >> 3;           // node row 0..15
  const int c0 = (threadIdx.x & 7) * 8;     // 8 cols per thread
  const int node = rt * 16 + r;
  float vv[8], s = 0.f;
#pragma unroll
  for (int i = 0; i < 8; ++i) {
    float v = tile[r][c0 + i] + b_p[c0 + i] + x[node * OUTD + c0 + i];
    v = (v > 0.f) ? v : (__expf(v) - 1.f);  // ELU
    vv[i] = v; s += v;
  }
  for (int m = 1; m < 8; m <<= 1) s += __shfl_xor(s, m, 8);
  const float mu = s * (1.f / OUTD);
  float q = 0.f;
#pragma unroll
  for (int i = 0; i < 8; ++i) { float dd = vv[i] - mu; q += dd * dd; }
  for (int m = 1; m < 8; m <<= 1) q += __shfl_xor(q, m, 8);
  const float inv = rsqrtf(q * (1.f / OUTD) + 1e-5f);
#pragma unroll
  for (int i = 0; i < 8; ++i)
    out[node * OUTD + c0 + i] = (vv[i] - mu) * inv * gamma[c0 + i] + beta[c0 + i];
}

// ---------------------------------------------------------------------------
extern "C" void kernel_launch(void* const* d_in, const int* in_sizes, int n_in,
                              void* d_out, int out_size, void* d_ws, size_t ws_size,
                              hipStream_t stream) {
  const float*     x         = (const float*)d_in[0];
  const long long* ei        = (const long long*)d_in[1];
  const float*     eattr     = (const float*)d_in[2];
  const float*     W_l       = (const float*)d_in[3];
  const float*     b_l       = (const float*)d_in[4];
  const float*     W_r       = (const float*)d_in[5];
  const float*     b_r       = (const float*)d_in[6];
  const float*     W_e       = (const float*)d_in[7];
  const float*     att       = (const float*)d_in[8];
  const float*     bias_conv = (const float*)d_in[9];
  const float*     W_p       = (const float*)d_in[10];
  const float*     b_p       = (const float*)d_in[11];
  const float*     gamma     = (const float*)d_in[12];
  const float*     beta      = (const float*)d_in[13];
  float*           out       = (float*)d_out;

  // workspace carve (~70 MB total; entire hot set stays L2-resident)
  char* base = (char*)d_ws; size_t off = 0;
  auto carve = [&](size_t bytes) -> void* {
    off = (off + 255) & ~(size_t)255;
    void* p = base + off; off += bytes; return p;
  };
  float*     eabuf   = (float*)carve(516 * sizeof(float)); // [0..511]=partials,[512]=mean
  _Float16*  xf16    = (_Float16*)carve((size_t)NN * IND * 2);
  _Float16*  bswz_lr = (_Float16*)carve((size_t)32768 * 2);
  _Float16*  bswz_p  = (_Float16*)carve((size_t)16384 * 2);
  float*     xl      = (float*)carve((size_t)NN * HD * 4);
  float*     xr      = (float*)carve((size_t)NN * HD * 4);
  float*     hagg    = (float*)carve((size_t)NN * HD * 4);
  float*     nodemax = (float*)carve((size_t)NN * HEADS * 4);
  float*     den     = (float*)carve((size_t)NN * HEADS * 4);
  float*     elog    = (float*)carve((size_t)ETOT * HEADS * 4);

  // 0) init accumulators
  k_init<<<(NN * HD + 255) / 256, 256, 0, stream>>>(hagg, den, nodemax, eabuf);
  // 1) deterministic edge_attr mean
  k_ea1<<<512, 256, 0, stream>>>(eattr, eabuf);
  k_ea2<<<1, 512, 0, stream>>>(eabuf);
  // 2) convert/swizzle operands for WMMA
  k_cvt_x<<<(NN * IND + 255) / 256, 256, 0, stream>>>(x, xf16);
  k_swz<<<192, 256, 0, stream>>>(W_l, W_r, W_p, bswz_lr, bswz_p);
  // 3) node transforms via WMMA
  k_gemm_lr<<<dim3(NN / 16, 8), 128, 0, stream>>>(xf16, bswz_lr, b_l, b_r, xl, xr);
  // 4) edge passes (wave per edge)
  k_logit<<<(ETOT + 7) / 8, 256, 0, stream>>>(ei, eattr, eabuf + 512, W_e, att,
                                              xl, xr, elog, nodemax);
  k_expsum<<<((size_t)ETOT * 4 + 255) / 256, 256, 0, stream>>>(ei, elog, nodemax, den);
  k_agg<<<(ETOT + 7) / 8, 256, 0, stream>>>(ei, elog, den, xl, hagg);
  // 5) projection + residual + ELU + LayerNorm via WMMA
  k_final<<<NN / 16, 128, 0, stream>>>(hagg, bias_conv, bswz_p, b_p, x,
                                       gamma, beta, out);
}
